// AesSA_56899726737881
// MI455X (gfx1250) — matrix operand order repaired and checked by reading, not compiled, not run on hardware
//
#include <hip/hip_runtime.h>
#include <math.h>

// ---------------------------------------------------------------------------
// Types / helpers
// ---------------------------------------------------------------------------
typedef __bf16 bf16_t;
typedef __attribute__((ext_vector_type(16))) __bf16 v16bf;
typedef __attribute__((ext_vector_type(8)))  float  v8f;

static __device__ __forceinline__ bf16_t f2bf(float f) {
    union { float f; unsigned u; } cv; cv.f = f;
    unsigned u = cv.u;
    u += 0x7FFFu + ((u >> 16) & 1u);          // round-to-nearest-even
    unsigned short h = (unsigned short)(u >> 16);
    union { unsigned short s; bf16_t b; } o; o.s = h;
    return o.b;
}
static __device__ __forceinline__ float bf2f(bf16_t b) {
    union { bf16_t b; unsigned short s; } i; i.b = b;
    union { unsigned u; float f; } o; o.u = ((unsigned)i.s) << 16;
    return o.f;
}
static __device__ __forceinline__ v8f vzero8() {
    v8f z;
#pragma unroll
    for (int i = 0; i < 8; ++i) z[i] = 0.f;
    return z;
}

// WMMA D = A*B + C  (16x16x32, bf16 in, f32 acc)
static __device__ __forceinline__ v8f wmma_bf(v16bf a, v16bf b, v8f c) {
    return __builtin_amdgcn_wmma_f32_16x16x32_bf16(
        false, a, false, b, (short)0, c, false, false);
}

// A-operand (16x32 bf16): lane row = lane&15 ; lanes<16 take K 0..7 & 16..23,
// lanes>=16 take K 8..15 & 24..31 (per CDNA5 ISA A layout).
static __device__ __forceinline__ v16bf load_A(const bf16_t* __restrict__ base,
                                               int rowStride, int kbase, int lane) {
    const bf16_t* p = base + (lane & 15) * rowStride + kbase + ((lane & 16) ? 8 : 0);
    v16bf a;
#pragma unroll
    for (int i = 0; i < 8; ++i) { a[i] = p[i]; a[8 + i] = p[16 + i]; }
    return a;
}

// B-operand (32x16 bf16): lane col = lane&15 ; lanes<16 hold K 0..7 & 16..23,
// lanes>=16 hold K 8..15 & 24..31 ; read from this lane's k-major row.
static __device__ __forceinline__ v16bf load_B(const bf16_t* __restrict__ rowBase,
                                               int kbase, int lane) {
    const bf16_t* p = rowBase + kbase + ((lane & 16) ? 8 : 0);
    v16bf b;
#pragma unroll
    for (int i = 0; i < 8; ++i) { b[i] = p[i]; b[8 + i] = p[16 + i]; }
    return b;
}

// ---------------------------------------------------------------------------
// Async direct-to-LDS staging (gfx1250 GLOBAL_LOAD_ASYNC_TO_LDS_B128, ASYNCcnt)
// with safe fallback to VGPR round-trip copies.
// Builtin prototype (from clang diagnostic):
//   void __builtin_amdgcn_global_load_async_to_lds_b128(
//        int4 addrspace(1)* src, int4 addrspace(3)* dst, Imm int off, Imm int cpol)
// ---------------------------------------------------------------------------
#if defined(__has_builtin)
#  if __has_builtin(__builtin_amdgcn_global_load_async_to_lds_b128) && \
      __has_builtin(__builtin_amdgcn_s_wait_asynccnt)
#    define USE_ASYNC_LDS 1
#  endif
#endif
#ifndef USE_ASYNC_LDS
#  define USE_ASYNC_LDS 0
#endif

typedef int v4i_t __attribute__((vector_size(16)));
typedef __attribute__((address_space(1))) v4i_t gv4_t;
typedef __attribute__((address_space(3))) v4i_t lv4_t;

// copy `bytes` (multiple of 16, both ptrs 16B aligned) global -> LDS, 256 thr
static __device__ __forceinline__ void stage_to_lds(bf16_t* __restrict__ dst,
                                                    const bf16_t* __restrict__ src,
                                                    int bytes, int tid) {
#if USE_ASYNC_LDS
    gv4_t* g = (gv4_t*)(void*)const_cast<bf16_t*>(src);
    lv4_t* l = (lv4_t*)(void*)dst;
    for (int i = tid; i < bytes / 16; i += 256)
        __builtin_amdgcn_global_load_async_to_lds_b128(g + i, l + i, 0, 0);
#else
    const uint4* s4 = (const uint4*)src;
    uint4* d4 = (uint4*)dst;
    for (int i = tid; i < bytes / 16; i += 256) d4[i] = s4[i];
#endif
}
static __device__ __forceinline__ void stage_fence() {
#if USE_ASYNC_LDS
    __builtin_amdgcn_s_wait_asynccnt(0);
#endif
}

// ---------------------------------------------------------------------------
// Problem constants
// ---------------------------------------------------------------------------
constexpr int CB = 4;     // batch
constexpr int CC = 512;   // channels
constexpr int CN = 4096;  // spatial (64*64)
constexpr int MB = 64;    // attention key-block
constexpr int VP = 72;    // padded LDS V row stride (elements, 144B rows)

// ---------------------------------------------------------------------------
// Small utility kernels
// ---------------------------------------------------------------------------
__global__ void cast_w_kernel(const float* __restrict__ in, bf16_t* __restrict__ out, int n) {
    int i = blockIdx.x * 256 + threadIdx.x;
    if (i < n) out[i] = f2bf(in[i]);
}

// per-(b,c) mean / rstd over N spatial elements
__global__ __launch_bounds__(256) void mvn_stats_kernel(const float* __restrict__ X,
                                                        float* __restrict__ stats, int N) {
    long row = blockIdx.x;
    const float* p = X + row * (long)N;
    float s = 0.f, s2 = 0.f;
    for (int i = threadIdx.x; i < N; i += 256) { float v = p[i]; s += v; s2 += v * v; }
    __shared__ float sh0[256], sh1[256];
    sh0[threadIdx.x] = s; sh1[threadIdx.x] = s2;
    __syncthreads();
    for (int off = 128; off > 0; off >>= 1) {
        if (threadIdx.x < off) { sh0[threadIdx.x] += sh0[threadIdx.x + off];
                                 sh1[threadIdx.x] += sh1[threadIdx.x + off]; }
        __syncthreads();
    }
    if (threadIdx.x == 0) {
        float mean = sh0[0] / N;
        float var  = sh1[0] / N - mean * mean;
        stats[row * 2 + 0] = mean;
        stats[row * 2 + 1] = rsqrtf(fmaxf(var, 0.f) + 1e-5f);
    }
}

// f32 [C,N] (per batch) -> bf16 [N,C] ; optional per-(b,c) normalization
__global__ __launch_bounds__(256) void cast_transpose_kernel(const float* __restrict__ In,
                                                             bf16_t* __restrict__ Out,
                                                             const float* __restrict__ stats,
                                                             int C, int N) {
    __shared__ float tile[32][33];
    int b  = blockIdx.z;
    int c0 = blockIdx.y * 32, n0 = blockIdx.x * 32;
    const float* inb = In + (long)b * C * N;
    for (int r = threadIdx.y; r < 32; r += 8) {
        int c = c0 + r;
        float v = inb[(long)c * N + n0 + threadIdx.x];
        if (stats) {
            float m  = stats[((long)b * C + c) * 2 + 0];
            float rs = stats[((long)b * C + c) * 2 + 1];
            v = (v - m) * rs;
        }
        tile[r][threadIdx.x] = v;
    }
    __syncthreads();
    bf16_t* outb = Out + (long)b * N * C;
    for (int r = threadIdx.y; r < 32; r += 8) {
        int n = n0 + r;
        outb[(long)n * C + c0 + threadIdx.x] = f2bf(tile[threadIdx.x][r]);
    }
}

// ---------------------------------------------------------------------------
// conv1x1 as WMMA GEMM:  Dt[b][n][o] = sum_i Xt[b][n][i] * W[o][i] + bias[o]
// (optional residual from f32 [C,N] layout, compile-time selected).
// ---------------------------------------------------------------------------
template <bool HAS_RES>
__global__ __launch_bounds__(256) void conv_gemm_kernel(const bf16_t* __restrict__ Xt,
                                                        const bf16_t* __restrict__ Wm,
                                                        const float* __restrict__ bias,
                                                        const float* __restrict__ resCN,
                                                        bf16_t* __restrict__ Out) {
    int lane = threadIdx.x & 31;
    int wave = threadIdx.x >> 5;
    long tile = (long)blockIdx.x * 8 + wave;
    const int tilesO = CC / 16;                 // 32
    const int tilesN = CN / 16;                 // 256
    const long tilesPerB = (long)tilesO * tilesN;
    long b = tile / tilesPerB;
    long t = tile % tilesPerB;
    int  to = (int)(t % tilesO);
    int  tn = (int)(t / tilesO);

    const bf16_t* Xb = Xt + (b * (long)CN + (long)tn * 16) * CC;
    const bf16_t* Wr = Wm + (long)(to * 16 + (lane & 15)) * CC;

    v8f acc = vzero8();
#pragma unroll 4
    for (int k = 0; k < CC; k += 32) {
        v16bf a  = load_A(Xb, CC, k, lane);
        v16bf bb = load_B(Wr, k, lane);
        acc = wmma_bf(a, bb, acc);
    }
    int   ocol  = to * 16 + (lane & 15);
    float bv    = bias[ocol];
    int   nbase = tn * 16 + ((lane & 16) ? 8 : 0);
    bf16_t* outp = Out + b * (long)CN * CC;
    const float* resb = HAS_RES ? (resCN + b * (long)CC * CN + (long)ocol * CN) : nullptr;
#pragma unroll
    for (int j = 0; j < 8; ++j) {
        int n = nbase + j;
        float v = acc[j] + bv;
        if (HAS_RES) v += resb[n];
        outp[(long)n * CC + ocol] = f2bf(v);
    }
}

// ---------------------------------------------------------------------------
// Attention 1 (sa_fusion core): O[n,c] = softmax_m(Q^T K)[n,m] * V^T[m,c]
// Flash-style online softmax; 8 waves split C, partial-S reduced in LDS.
// ---------------------------------------------------------------------------
__global__ __launch_bounds__(256) void attention1_kernel(const bf16_t* __restrict__ qT,
                                                         const bf16_t* __restrict__ kT,
                                                         const bf16_t* __restrict__ vT,
                                                         bf16_t* __restrict__ outT) {
    extern __shared__ char smem[];
    bf16_t* ldsQ = (bf16_t*)smem;                    // 16*CC
    bf16_t* ldsK = ldsQ + 16 * CC;                   // MB*CC
    bf16_t* ldsV = ldsK + MB * CC;                   // CC*VP (transposed, padded)
    bf16_t* ldsP = ldsV + CC * VP;                   // 16*MB
    float*  ldsSp = (float*)(ldsP + 16 * MB);        // 8*16*MB
    float*  ldsS  = ldsSp + 8 * 16 * MB;             // 16*MB
    float*  rowMax   = ldsS + 16 * MB;
    float*  rowSum   = rowMax + 16;
    float*  rowAlpha = rowSum + 16;

    int tid = threadIdx.x, lane = tid & 31, wave = tid >> 5;
    int b  = blockIdx.y;
    int n0 = blockIdx.x * 16;
    const bf16_t* Qb = qT + ((long)b * CN + n0) * CC;
    const bf16_t* Kb = kT + (long)b * CN * CC;
    const bf16_t* Vb = vT + (long)b * CN * CC;

    stage_to_lds(ldsQ, Qb, 16 * CC * 2, tid);
    if (tid < 16) { rowMax[tid] = -3.0e38f; rowSum[tid] = 0.f; }
    stage_fence();
    __syncthreads();

    int cw0 = wave * 64;
    // Q A-fragments are invariant over the whole m-loop: hoist.
    v16bf aQ0 = load_A(ldsQ, CC, cw0 +  0, lane);
    v16bf aQ1 = load_A(ldsQ, CC, cw0 + 32, lane);

    v8f O[4];
#pragma unroll
    for (int t = 0; t < 4; ++t) O[t] = vzero8();

    for (int m0 = 0; m0 < CN; m0 += MB) {
        __syncthreads();
        stage_to_lds(ldsK, Kb + (long)m0 * CC, MB * CC * 2, tid);
        // stage V tile transposed: ldsV[c*VP + m] = vT[m0+m][c]; pack 2 m's/b32
        for (int idx = tid; idx < (MB / 2) * (CC / 8); idx += 256) {
            int m2 = (idx / (CC / 8)) * 2;
            int c8 = (idx % (CC / 8)) * 8;
            const bf16_t* p0 = Vb + (long)(m0 + m2) * CC + c8;
            const bf16_t* p1 = p0 + CC;
#pragma unroll
            for (int i = 0; i < 8; ++i) {
                union { bf16_t h[2]; unsigned u; } pk;
                pk.h[0] = p0[i]; pk.h[1] = p1[i];
                *(unsigned*)&ldsV[(c8 + i) * VP + m2] = pk.u;
            }
        }
        stage_fence();
        __syncthreads();
        // partial S over this wave's 64 channels
#pragma unroll
        for (int mt = 0; mt < 4; ++mt) {
            const bf16_t* krow = ldsK + (mt * 16 + (lane & 15)) * CC;
            v8f s = vzero8();
            s = wmma_bf(aQ0, load_B(krow, cw0 +  0, lane), s);
            s = wmma_bf(aQ1, load_B(krow, cw0 + 32, lane), s);
            int col = mt * 16 + (lane & 15);
            int nb  = (lane & 16) ? 8 : 0;
#pragma unroll
            for (int j = 0; j < 8; ++j) ldsSp[(wave * 16 + nb + j) * MB + col] = s[j];
        }
        __syncthreads();
        // reduce partials
        for (int idx = tid; idx < 16 * MB; idx += 256) {
            int n = idx / MB, m = idx % MB;
            float acc = 0.f;
#pragma unroll
            for (int w = 0; w < 8; ++w) acc += ldsSp[(w * 16 + n) * MB + m];
            ldsS[idx] = acc;
        }
        __syncthreads();
        // online softmax stats per row
        if (tid < 16) {
            const float* r = ldsS + tid * MB;
            float mOld = rowMax[tid], mNew = mOld;
            for (int m = 0; m < MB; ++m) mNew = fmaxf(mNew, r[m]);
            float alpha = __expf(mOld - mNew);
            float add = 0.f;
            for (int m = 0; m < MB; ++m) add += __expf(r[m] - mNew);
            rowSum[tid] = rowSum[tid] * alpha + add;
            rowMax[tid] = mNew;
            rowAlpha[tid] = alpha;
        }
        __syncthreads();
        for (int idx = tid; idx < 16 * MB; idx += 256) {
            int n = idx / MB;
            ldsP[idx] = f2bf(__expf(ldsS[idx] - rowMax[n]));
        }
        __syncthreads();
        // rescale + accumulate O
        float av[8];
        { int nb = (lane & 16) ? 8 : 0;
#pragma unroll
          for (int j = 0; j < 8; ++j) av[j] = rowAlpha[nb + j]; }
#pragma unroll
        for (int ct = 0; ct < 4; ++ct) {
#pragma unroll
            for (int j = 0; j < 8; ++j) O[ct][j] *= av[j];
            int ccol = cw0 + ct * 16 + (lane & 15);
#pragma unroll
            for (int kk = 0; kk < 2; ++kk) {
                v16bf a  = load_A(ldsP, MB, kk * 32, lane);
                v16bf bb = load_B(ldsV + ccol * VP, kk * 32, lane);
                O[ct] = wmma_bf(a, bb, O[ct]);
            }
        }
    }
    __syncthreads();
    if (tid < 16) rowAlpha[tid] = 1.0f / rowSum[tid];
    __syncthreads();
    float inv[8];
    { int nb = (lane & 16) ? 8 : 0;
#pragma unroll
      for (int j = 0; j < 8; ++j) inv[j] = rowAlpha[nb + j]; }
    bf16_t* outp = outT + ((long)b * CN + n0) * CC;
#pragma unroll
    for (int ct = 0; ct < 4; ++ct) {
        int ccol = cw0 + ct * 16 + (lane & 15);
        int nb = (lane & 16) ? 8 : 0;
#pragma unroll
        for (int j = 0; j < 8; ++j)
            outp[(long)(nb + j) * CC + ccol] = f2bf(O[ct][j] * inv[j]);
    }
}

// ---------------------------------------------------------------------------
// Attention 2 (AdaAttN core) with fused epilogue:
//   S = softmax(Fq^T Gk) ; mean = S Hv^T ; second = S (Hv^2)^T
//   out = sqrt(relu(second-mean^2)) * mvn(content) + mean   (f32 [B,C,N])
// ---------------------------------------------------------------------------
__global__ __launch_bounds__(256) void attention2_kernel(const bf16_t* __restrict__ fqT,
                                                         const bf16_t* __restrict__ gkT,
                                                         const bf16_t* __restrict__ hvT,
                                                         const float* __restrict__ content,
                                                         const float* __restrict__ stats,
                                                         float* __restrict__ out) {
    extern __shared__ char smem[];
    bf16_t* ldsQ  = (bf16_t*)smem;
    bf16_t* ldsK  = ldsQ + 16 * CC;
    bf16_t* ldsV  = ldsK + MB * CC;
    bf16_t* ldsV2 = ldsV + CC * VP;
    bf16_t* ldsP  = ldsV2 + CC * VP;
    float*  ldsSp = (float*)(ldsP + 16 * MB);
    float*  ldsS  = ldsSp + 8 * 16 * MB;
    float*  rowMax   = ldsS + 16 * MB;
    float*  rowSum   = rowMax + 16;
    float*  rowAlpha = rowSum + 16;

    int tid = threadIdx.x, lane = tid & 31, wave = tid >> 5;
    int b  = blockIdx.y;
    int n0 = blockIdx.x * 16;
    const bf16_t* Qb = fqT + ((long)b * CN + n0) * CC;
    const bf16_t* Kb = gkT + (long)b * CN * CC;
    const bf16_t* Vb = hvT + (long)b * CN * CC;

    stage_to_lds(ldsQ, Qb, 16 * CC * 2, tid);
    if (tid < 16) { rowMax[tid] = -3.0e38f; rowSum[tid] = 0.f; }
    stage_fence();
    __syncthreads();

    int cw0 = wave * 64;
    v16bf aQ0 = load_A(ldsQ, CC, cw0 +  0, lane);
    v16bf aQ1 = load_A(ldsQ, CC, cw0 + 32, lane);

    v8f O[4], O2[4];
#pragma unroll
    for (int t = 0; t < 4; ++t) { O[t] = vzero8(); O2[t] = vzero8(); }

    for (int m0 = 0; m0 < CN; m0 += MB) {
        __syncthreads();
        stage_to_lds(ldsK, Kb + (long)m0 * CC, MB * CC * 2, tid);
        for (int idx = tid; idx < (MB / 2) * (CC / 8); idx += 256) {
            int m2 = (idx / (CC / 8)) * 2;
            int c8 = (idx % (CC / 8)) * 8;
            const bf16_t* p0 = Vb + (long)(m0 + m2) * CC + c8;
            const bf16_t* p1 = p0 + CC;
#pragma unroll
            for (int i = 0; i < 8; ++i) {
                float f0 = bf2f(p0[i]), f1 = bf2f(p1[i]);
                union { bf16_t h[2]; unsigned u; } pk, pk2;
                pk.h[0]  = p0[i];          pk.h[1]  = p1[i];
                pk2.h[0] = f2bf(f0 * f0);  pk2.h[1] = f2bf(f1 * f1);
                *(unsigned*)&ldsV [(c8 + i) * VP + m2] = pk.u;
                *(unsigned*)&ldsV2[(c8 + i) * VP + m2] = pk2.u;
            }
        }
        stage_fence();
        __syncthreads();
#pragma unroll
        for (int mt = 0; mt < 4; ++mt) {
            const bf16_t* krow = ldsK + (mt * 16 + (lane & 15)) * CC;
            v8f s = vzero8();
            s = wmma_bf(aQ0, load_B(krow, cw0 +  0, lane), s);
            s = wmma_bf(aQ1, load_B(krow, cw0 + 32, lane), s);
            int col = mt * 16 + (lane & 15);
            int nb  = (lane & 16) ? 8 : 0;
#pragma unroll
            for (int j = 0; j < 8; ++j) ldsSp[(wave * 16 + nb + j) * MB + col] = s[j];
        }
        __syncthreads();
        for (int idx = tid; idx < 16 * MB; idx += 256) {
            int n = idx / MB, m = idx % MB;
            float acc = 0.f;
#pragma unroll
            for (int w = 0; w < 8; ++w) acc += ldsSp[(w * 16 + n) * MB + m];
            ldsS[idx] = acc;
        }
        __syncthreads();
        if (tid < 16) {
            const float* r = ldsS + tid * MB;
            float mOld = rowMax[tid], mNew = mOld;
            for (int m = 0; m < MB; ++m) mNew = fmaxf(mNew, r[m]);
            float alpha = __expf(mOld - mNew);
            float add = 0.f;
            for (int m = 0; m < MB; ++m) add += __expf(r[m] - mNew);
            rowSum[tid] = rowSum[tid] * alpha + add;
            rowMax[tid] = mNew;
            rowAlpha[tid] = alpha;
        }
        __syncthreads();
        for (int idx = tid; idx < 16 * MB; idx += 256) {
            int n = idx / MB;
            ldsP[idx] = f2bf(__expf(ldsS[idx] - rowMax[n]));
        }
        __syncthreads();
        float av[8];
        { int nb = (lane & 16) ? 8 : 0;
#pragma unroll
          for (int j = 0; j < 8; ++j) av[j] = rowAlpha[nb + j]; }
#pragma unroll
        for (int ct = 0; ct < 4; ++ct) {
#pragma unroll
            for (int j = 0; j < 8; ++j) { O[ct][j] *= av[j]; O2[ct][j] *= av[j]; }
            int ccol = cw0 + ct * 16 + (lane & 15);
#pragma unroll
            for (int kk = 0; kk < 2; ++kk) {
                v16bf a   = load_A(ldsP, MB, kk * 32, lane);
                v16bf bb  = load_B(ldsV  + ccol * VP, kk * 32, lane);
                v16bf bb2 = load_B(ldsV2 + ccol * VP, kk * 32, lane);
                O[ct]  = wmma_bf(a, bb,  O[ct]);
                O2[ct] = wmma_bf(a, bb2, O2[ct]);
            }
        }
    }
    __syncthreads();
    if (tid < 16) rowAlpha[tid] = 1.0f / rowSum[tid];
    __syncthreads();
    float inv[8];
    { int nb = (lane & 16) ? 8 : 0;
#pragma unroll
      for (int j = 0; j < 8; ++j) inv[j] = rowAlpha[nb + j]; }

    const float* cb = content + (long)b * CC * CN;
    float* ob = out + (long)b * CC * CN;
#pragma unroll
    for (int ct = 0; ct < 4; ++ct) {
        int c = cw0 + ct * 16 + (lane & 15);
        float mu = stats[((long)b * CC + c) * 2 + 0];
        float rs = stats[((long)b * CC + c) * 2 + 1];
        int nb = (lane & 16) ? 8 : 0;
#pragma unroll
        for (int j = 0; j < 8; ++j) {
            int n = n0 + nb + j;
            float mean = O[ct][j]  * inv[j];
            float sec  = O2[ct][j] * inv[j];
            float sd = sqrtf(fmaxf(sec - mean * mean, 0.f));
            float ck = (cb[(long)c * CN + n] - mu) * rs;
            ob[(long)c * CN + n] = sd * ck + mean;
        }
    }
}

// ---------------------------------------------------------------------------
// Host launch
// ---------------------------------------------------------------------------
extern "C" void kernel_launch(void* const* d_in, const int* in_sizes, int n_in,
                              void* d_out, int out_size, void* d_ws, size_t ws_size,
                              hipStream_t stream) {
    (void)in_sizes; (void)n_in; (void)out_size; (void)ws_size;
    const float* content = (const float*)d_in[0];
    const float* style   = (const float*)d_in[1];
    const float* aes     = (const float*)d_in[2];
    const float* Wraw[7] = { (const float*)d_in[3], (const float*)d_in[5],
                             (const float*)d_in[7], (const float*)d_in[9],
                             (const float*)d_in[11], (const float*)d_in[13],
                             (const float*)d_in[15] };                 // Wf1 Wf2 Wf3 Wfrs Wf Wg Wh
    const float* bias[7] = { (const float*)d_in[4], (const float*)d_in[6],
                             (const float*)d_in[8], (const float*)d_in[10],
                             (const float*)d_in[12], (const float*)d_in[14],
                             (const float*)d_in[16] };                 // bf1 bf2 bf3 bfrs bf bg bh
    float* out = (float*)d_out;

    char* ws = (char*)d_ws;
    size_t off = 0;
    auto carve = [&](size_t bytes) { void* p = ws + off; off += (bytes + 255) & ~(size_t)255; return p; };

    bf16_t* wb[7];
    for (int i = 0; i < 7; ++i) wb[i] = (bf16_t*)carve((size_t)CC * CC * sizeof(bf16_t));
    float* stats = (float*)carve((size_t)CB * CC * 2 * sizeof(float));
    const size_t BUF = (size_t)CB * CN * CC * sizeof(bf16_t);
    bf16_t* sT    = (bf16_t*)carve(BUF);  // style^T
    bf16_t* aT    = (bf16_t*)carve(BUF);  // aes^T    (reused: Fq^T)
    bf16_t* ckT   = (bf16_t*)carve(BUF);  // mvn(content)^T
    bf16_t* qT    = (bf16_t*)carve(BUF);  // q^T      (reused: style_key^T)
    bf16_t* kTb   = (bf16_t*)carve(BUF);  // k^T      (reused: Gk^T)
    bf16_t* vTb   = (bf16_t*)carve(BUF);  // v^T      (reused: Hv^T)
    bf16_t* xfrsT = (bf16_t*)carve(BUF);  // attention-1 output

    // 1. weights -> bf16
    const int wn = CC * CC;
    for (int i = 0; i < 7; ++i)
        cast_w_kernel<<<(wn + 255) / 256, 256, 0, stream>>>(Wraw[i], wb[i], wn);

    // 2. content stats (mvn)
    mvn_stats_kernel<<<CB * CC, 256, 0, stream>>>(content, stats, CN);

    // 3. activation transposes -> bf16 [B,N,C]
    dim3 tg(CN / 32, CC / 32, CB), tb(32, 8);
    cast_transpose_kernel<<<tg, tb, 0, stream>>>(style,   sT,  nullptr, CC, CN);
    cast_transpose_kernel<<<tg, tb, 0, stream>>>(aes,     aT,  nullptr, CC, CN);
    cast_transpose_kernel<<<tg, tb, 0, stream>>>(content, ckT, stats,   CC, CN);

    // 4. conv1x1 q/k/v (WMMA GEMMs)
    const int convBlocks = (CB * (CN / 16) * (CC / 16)) / 8;
    conv_gemm_kernel<false><<<convBlocks, 256, 0, stream>>>(sT, wb[0], bias[0], nullptr, qT);
    conv_gemm_kernel<false><<<convBlocks, 256, 0, stream>>>(aT, wb[1], bias[1], nullptr, kTb);
    conv_gemm_kernel<false><<<convBlocks, 256, 0, stream>>>(aT, wb[2], bias[2], nullptr, vTb);

    // 5. attention 1 (fused softmax, flash-style)
    size_t smem1 = (size_t)(16 * CC + MB * CC + CC * VP + 16 * MB) * sizeof(bf16_t)
                 + (size_t)(8 * 16 * MB + 16 * MB + 64) * sizeof(float);
    attention1_kernel<<<dim3(CN / 16, CB), 256, smem1, stream>>>(qT, kTb, vTb, xfrsT);

    // 6. style_key = Wfrs*x_frs + bfrs + style ; then Fq / Gk / Hv
    conv_gemm_kernel<true ><<<convBlocks, 256, 0, stream>>>(xfrsT, wb[3], bias[3], style, qT);   // style_key^T
    conv_gemm_kernel<false><<<convBlocks, 256, 0, stream>>>(ckT,   wb[4], bias[4], nullptr, aT); // Fq^T
    conv_gemm_kernel<false><<<convBlocks, 256, 0, stream>>>(qT,    wb[5], bias[5], nullptr, kTb);// Gk^T
    conv_gemm_kernel<false><<<convBlocks, 256, 0, stream>>>(sT,    wb[6], bias[6], nullptr, vTb);// Hv^T

    // 7. attention 2 with fused AdaAttN epilogue
    size_t smem2 = smem1 + (size_t)(CC * VP) * sizeof(bf16_t);
    attention2_kernel<<<dim3(CN / 16, CB), 256, smem2, stream>>>(aT, kTb, vTb, content, stats, out);
}
